// ExomaAttention_21406117003536
// MI455X (gfx1250) — matrix operand order repaired and checked by reading, not compiled, not run on hardware
//
#include <hip/hip_runtime.h>
#include <hip/hip_bf16.h>
#include <stdint.h>

// Problem constants (reference: B=32, T=128, H=4096, NH=32, NKV=8, HD=128, S=128)
#define NHEAD   32
#define NKVH    8
#define HDIM    128
#define BATCH   32
#define SEQ     128
#define HID     4096
#define QKVN    6144   // NH*HD + 2*NKV*HD
#define KOFF    4096   // start of K block in qkv row
#define VOFF    5120   // start of V block in qkv row

typedef __attribute__((ext_vector_type(16))) _Float16 v16h;
typedef __attribute__((ext_vector_type(8)))  float    v8f;

union FragH { v16h v; uint32_t u[8]; };
union FragF { v8f  v; float    f[8]; };

// ---------------------------------------------------------------------------
// CDNA5 async global->LDS copy (GLOBAL_LOAD_ASYNC_TO_LDS_B128, GV mode,
// tracked by ASYNCcnt). LDS address = low 32 bits of the generic pointer
// (aperture scheme).
// ---------------------------------------------------------------------------
__device__ __forceinline__ void async_copy16(void* lds_dst, const void* gsrc) {
  const uint32_t l = (uint32_t)(uintptr_t)lds_dst;
  asm volatile("global_load_async_to_lds_b128 %0, %1, off"
               :: "v"(l), "v"(gsrc)
               : "memory");
}
__device__ __forceinline__ void wait_async0() {
  asm volatile("s_wait_asynccnt 0x0" ::: "memory");
}

// ---------------------------------------------------------------------------
// WMMA fragment helpers (wave32, V_WMMA_F32_16X16X32_F16)
// A (16x32 f16): lane m = lane&15; VGPR j holds K = (j/4)*16 + (lane>=16?8:0) + 2*(j&3) {+0,+1}
// B (32x16 f16), n-major LDS: lane n = lane&15; VGPR j holds K = (j/4)*16 + (lane>=16?8:0) + 2*(j&3) {+0,+1}
// C/D (16x16 f32): VGPR v holds M = v + (lane>=16?8:0), N = lane&15 (+16 per n-tile)
// ---------------------------------------------------------------------------
__device__ __forceinline__ v16h load_a_frag(const _Float16* lds, int row0, int stride,
                                            int kbase, int lane) {
  const int hfl = (lane >> 4) & 1;
  const int m   = lane & 15;
  FragH f;
  const _Float16* rp = lds + (row0 + m) * stride + kbase;
#pragma unroll
  for (int j = 0; j < 8; ++j) {
    const int k = ((j >> 2) << 4) + (hfl << 3) + ((j & 3) << 1);
    f.u[j] = *reinterpret_cast<const uint32_t*>(rp + k);
  }
  return f.v;
}

__device__ __forceinline__ v16h load_b_frag(const _Float16* lds, int col0, int stride,
                                            int kbase, int lane) {
  const int hfl = (lane >> 4) & 1;
  const int n   = lane & 15;
  FragH f;
  const _Float16* cp = lds + (col0 + n) * stride + kbase;
#pragma unroll
  for (int j = 0; j < 8; ++j)
    f.u[j] = *reinterpret_cast<const uint32_t*>(cp + ((j >> 2) << 4) + (hfl << 3) + ((j & 3) << 1));
  return f.v;
}

__device__ __forceinline__ v8f wmma_f16(v16h a, v16h b, v8f c) {
  // 8 args: (neg_a, A, neg_b, B, c_mod, C, reuse_a, reuse_b)
  return __builtin_amdgcn_wmma_f32_16x16x32_f16(false, a, false, b, (short)0, c,
                                                false, false);
}

// ---------------------------------------------------------------------------
// f32 -> f16 converters (plain + transposing; transposed output makes every
// GEMM B-operand n-major so B fragments are aligned u32 LDS loads)
// ---------------------------------------------------------------------------
__global__ void cvt_f32_f16_kernel(const float* __restrict__ s, _Float16* __restrict__ d,
                                   int n) {
  int i = blockIdx.x * blockDim.x + threadIdx.x;
  const int stride = gridDim.x * blockDim.x;
  for (; i < n; i += stride) d[i] = (_Float16)s[i];
}

__global__ void cvtT_f32_f16_kernel(const float* __restrict__ s, _Float16* __restrict__ d,
                                    int R, int C) {
  int i = blockIdx.x * blockDim.x + threadIdx.x;
  const int stride = gridDim.x * blockDim.x;
  const int n = R * C;
  for (; i < n; i += stride) {
    const int r = i / C;
    const int c = i - r * C;
    d[(size_t)c * R + r] = (_Float16)s[i];
  }
}

// ---------------------------------------------------------------------------
// Tiled WMMA GEMM: C[M,N] = A[M,K] (row-major f16) * B (given as Bt[N,K] f16).
// Compile-time shapes -> immediate-offset epilogue stores, no runtime branch.
// Block = 256 threads (8 waves), tile 128x128, BK=32.
// Wave w -> 32x64 sub-block = 2x4 WMMA tiles. Tile fill via async->LDS DMA.
// ---------------------------------------------------------------------------
template <int M, int N, int K, bool F32OUT>
__global__ __launch_bounds__(256) void gemm_f16_wmma(const _Float16* __restrict__ A,
                                                     const _Float16* __restrict__ Bt,
                                                     void* __restrict__ C) {
  __shared__ _Float16 As[128 * 32];
  __shared__ _Float16 Bs[128 * 32];
  const int tid  = threadIdx.x;
  const int lane = tid & 31;
  const int w    = tid >> 5;
  const int m0   = blockIdx.y * 128;
  const int n0   = blockIdx.x * 128;
  const int wm   = (w >> 1) << 5;   // 0,32,64,96
  const int wn   = (w & 1) << 6;    // 0,64

  FragF acc[2][4];
#pragma unroll
  for (int mt = 0; mt < 2; ++mt)
#pragma unroll
    for (int nt = 0; nt < 4; ++nt)
#pragma unroll
      for (int v = 0; v < 8; ++v) acc[mt][nt].f[v] = 0.0f;

  const int r  = tid >> 1;
  const int c0 = (tid & 1) << 4;  // 0 or 16
  const _Float16* gA = A  + (size_t)(m0 + r) * K + c0;
  const _Float16* gB = Bt + (size_t)(n0 + r) * K + c0;
  _Float16* lA = As + r * 32 + c0;
  _Float16* lB = Bs + r * 32 + c0;

#pragma unroll 1
  for (int k0 = 0; k0 < K; k0 += 32) {
    async_copy16(lA,     gA + k0);
    async_copy16(lA + 8, gA + k0 + 8);
    async_copy16(lB,     gB + k0);
    async_copy16(lB + 8, gB + k0 + 8);
    if (k0 + 32 < K) {
      __builtin_prefetch(gA + k0 + 32, 0, 1);
      __builtin_prefetch(gB + k0 + 32, 0, 1);
    }
    wait_async0();
    __syncthreads();

    v16h af[2];
#pragma unroll
    for (int mt = 0; mt < 2; ++mt) af[mt] = load_a_frag(As, wm + (mt << 4), 32, 0, lane);
#pragma unroll
    for (int nt = 0; nt < 4; ++nt) {
      const v16h bf = load_b_frag(Bs, wn + (nt << 4), 32, 0, lane);
#pragma unroll
      for (int mt = 0; mt < 2; ++mt)
        acc[mt][nt].v = wmma_f16(af[mt], bf, acc[mt][nt].v);
    }
    __syncthreads();
  }

  const int hfl = (lane >> 4) & 1;
  const int nl  = lane & 15;
#pragma unroll
  for (int mt = 0; mt < 2; ++mt)
#pragma unroll
    for (int nt = 0; nt < 4; ++nt) {
      const int gr = m0 + wm + (mt << 4) + (hfl << 3);
      const int gc = n0 + wn + (nt << 4) + nl;
      if constexpr (F32OUT) {
        float* cp = reinterpret_cast<float*>(C) + (size_t)gr * N + gc;
#pragma unroll
        for (int v = 0; v < 8; ++v) cp[(size_t)v * N] = acc[mt][nt].f[v];
      } else {
        _Float16* cp = reinterpret_cast<_Float16*>(C) + (size_t)gr * N + gc;
#pragma unroll
        for (int v = 0; v < 8; ++v) cp[(size_t)v * N] = (_Float16)acc[mt][nt].f[v];
      }
    }
}

// ---------------------------------------------------------------------------
// Fused per-head DCT attention. grid.x = B*NH. Block = 256 threads (8 waves).
// Each wave owns 16 full output rows (8 accumulator tiles), so row-softmax is
// done in registers with half-wave __shfl_xor reductions.
// LDS: X (A operand, row-major) + Y (B operand, n-major) = 64 KB.
// query is staged to global (wsq) and reloaded transposed for stage 4.
// ---------------------------------------------------------------------------
__global__ __launch_bounds__(256) void dct_attn_kernel(const _Float16* __restrict__ qkv,
                                                       const _Float16* __restrict__ projT,
                                                       _Float16* __restrict__ wsq,
                                                       _Float16* __restrict__ attn) {
  __shared__ _Float16 X[128 * 128];
  __shared__ _Float16 Y[128 * 128];
  const int tid  = threadIdx.x;
  const int lane = tid & 31;
  const int w    = tid >> 5;
  const int bh   = blockIdx.x;
  const int b    = bh >> 5;
  const int h    = bh & 31;
  const int kvh  = h >> 2;          // GQA: 4 query heads per kv head
  const int row0 = w << 4;
  const int hfl  = (lane >> 4) & 1;
  const int nl   = lane & 15;
  const _Float16* qkvB = qkv + (size_t)b * SEQ * QKVN;
  _Float16* wq = wsq + (size_t)bh * (HDIM * HDIM);

  FragF acc[8];

  auto mm = [&]() {   // acc[j] (j = n-tile) += X[row0..row0+16, :128] * Y(n-major)
#pragma unroll
    for (int j = 0; j < 8; ++j)
#pragma unroll
      for (int v = 0; v < 8; ++v) acc[j].f[v] = 0.0f;
#pragma unroll
    for (int kb = 0; kb < 128; kb += 32) {
      const v16h a = load_a_frag(X, row0, 128, kb, lane);
#pragma unroll
      for (int j = 0; j < 8; ++j) {
        const v16h bf = load_b_frag(Y, j << 4, 128, kb, lane);
        acc[j].v = wmma_f16(a, bf, acc[j].v);
      }
    }
  };

  auto softmax_rows = [&]() {  // softmax over N (128) for each of the wave's 16 rows
#pragma unroll
    for (int v = 0; v < 8; ++v) {
      float mx = -3.0e38f;
#pragma unroll
      for (int j = 0; j < 8; ++j) mx = fmaxf(mx, acc[j].f[v]);
#pragma unroll
      for (int s = 1; s < 16; s <<= 1) mx = fmaxf(mx, __shfl_xor(mx, s, 32));
      float sum = 0.0f;
#pragma unroll
      for (int j = 0; j < 8; ++j) {
        const float e = __expf(acc[j].f[v] - mx);
        acc[j].f[v] = e;
        sum += e;
      }
#pragma unroll
      for (int s = 1; s < 16; s <<= 1) sum += __shfl_xor(sum, s, 32);
      const float inv = 1.0f / sum;
#pragma unroll
      for (int j = 0; j < 8; ++j) acc[j].f[v] *= inv;
    }
  };

  // -- async fill: X <- q head tile (row-major), Y <- projT ([e][s], n-major)
  {
    const int r  = tid >> 1;
    const int c0 = (tid & 1) << 6;
    const _Float16* gq = qkvB + (size_t)r * QKVN + h * HDIM + c0;
    const _Float16* gp = projT + r * HDIM + c0;
    _Float16* xd = X + r * 128 + c0;
    _Float16* yd = Y + r * 128 + c0;
#pragma unroll
    for (int i = 0; i < 8; ++i) {
      async_copy16(xd + i * 8, gq + i * 8);
      async_copy16(yd + i * 8, gp + i * 8);
    }
    wait_async0();
  }
  __syncthreads();

  // -- stage 1: query = softmax(q @ proj), spill row-major to global staging
  mm();
  softmax_rows();
#pragma unroll
  for (int j = 0; j < 8; ++j)
#pragma unroll
    for (int v = 0; v < 8; ++v) {
      const int m = row0 + v + (hfl << 3);
      const int n = nl + (j << 4);
      wq[m * HDIM + n] = (_Float16)acc[j].f[v];
    }
  __threadfence();
  __syncthreads();

  // -- X <- k^T (A operand for keyp): transpose during LDS fill
  {
    const int s  = tid >> 1;
    const int d0 = (tid & 1) << 6;
    const _Float16* kr = qkvB + (size_t)s * QKVN + KOFF + kvh * HDIM + d0;
#pragma unroll
    for (int i = 0; i < 64; ++i) X[(d0 + i) * 128 + s] = kr[i];
  }
  __syncthreads();

  // -- stage 2: keyp = softmax(k^T @ proj); write in place to own X rows
  mm();
  softmax_rows();
#pragma unroll
  for (int j = 0; j < 8; ++j)
#pragma unroll
    for (int v = 0; v < 8; ++v) {
      const int m = row0 + v + (hfl << 3);
      const int n = nl + (j << 4);
      X[m * 128 + n] = (_Float16)acc[j].f[v];
    }
  __syncthreads();

  // -- Y <- v^T (n-major B for scores = keyp @ v)
  {
    const int e  = tid >> 1;
    const int f0 = (tid & 1) << 6;
    const _Float16* vr = qkvB + (size_t)e * QKVN + VOFF + kvh * HDIM + f0;
#pragma unroll
    for (int i = 0; i < 64; ++i) Y[(f0 + i) * 128 + e] = vr[i];
  }
  __syncthreads();

  // -- stage 3: scores = keyp @ v; in place to own X rows
  mm();
#pragma unroll
  for (int j = 0; j < 8; ++j)
#pragma unroll
    for (int v = 0; v < 8; ++v) {
      const int m = row0 + v + (hfl << 3);
      const int n = nl + (j << 4);
      X[m * 128 + n] = (_Float16)acc[j].f[v];
    }
  __syncthreads();

  // -- Y <- query^T (n-major B for out = scores @ query)
  {
    const int f  = tid >> 1;
    const int g0 = (tid & 1) << 6;
    const _Float16* qr = wq + f * HDIM + g0;
#pragma unroll
    for (int i = 0; i < 64; ++i) Y[(g0 + i) * 128 + f] = qr[i];
  }
  __syncthreads();

  // -- stage 4: out = scores @ query -> attn[b, t=d, h*128 + g]
  mm();
#pragma unroll
  for (int j = 0; j < 8; ++j)
#pragma unroll
    for (int v = 0; v < 8; ++v) {
      const int m = row0 + v + (hfl << 3);  // token index d
      const int n = nl + (j << 4);          // g
      attn[((size_t)b * SEQ + m) * HID + h * HDIM + n] = (_Float16)acc[j].f[v];
    }
}

// ---------------------------------------------------------------------------
// Host launcher. Inputs: 0=hidden_states 1=k_cache 2=v_cache 3=mask
// 4=qkv_w 5=o_w 6=proj 7=kv_write_indices.
// kv_write_indices == arange(T) with T == S, so caches are fully overwritten
// and k_cache/v_cache/mask/indices are dead inputs.
// ---------------------------------------------------------------------------
extern "C" void kernel_launch(void* const* d_in, const int* in_sizes, int n_in,
                              void* d_out, int out_size, void* d_ws, size_t ws_size,
                              hipStream_t stream) {
  (void)in_sizes; (void)n_in; (void)out_size; (void)ws_size;
  const float* hs    = (const float*)d_in[0];
  const float* qkv_w = (const float*)d_in[4];
  const float* o_w   = (const float*)d_in[5];
  const float* proj  = (const float*)d_in[6];

  char* p = (char*)d_ws;
  auto take = [&](size_t bytes) {
    char* q = p;
    p += (bytes + 255) & ~(size_t)255;
    return q;
  };
  _Float16* hs16   = (_Float16*)take((size_t)4096 * 4096 * 2); // hidden f16
  _Float16* qkvwT  = (_Float16*)take((size_t)6144 * 4096 * 2); // qkv_w^T f16
  _Float16* owT    = (_Float16*)take((size_t)4096 * 4096 * 2); // o_w^T f16
  _Float16* projT  = (_Float16*)take((size_t)128 * 128 * 2);   // proj^T f16
  _Float16* qkv16  = (_Float16*)take((size_t)4096 * 6144 * 2); // qkv activations
  _Float16* attn16 = (_Float16*)take((size_t)4096 * 4096 * 2); // attn output
  _Float16* wsq    = (_Float16*)take((size_t)1024 * 128 * 128 * 2); // query staging

  cvt_f32_f16_kernel<<<dim3(1024), dim3(256), 0, stream>>>(hs, hs16, 4096 * 4096);
  cvtT_f32_f16_kernel<<<dim3(1024), dim3(256), 0, stream>>>(qkv_w, qkvwT, 4096, 6144);
  cvtT_f32_f16_kernel<<<dim3(1024), dim3(256), 0, stream>>>(o_w, owT, 4096, 4096);
  cvtT_f32_f16_kernel<<<dim3(64), dim3(256), 0, stream>>>(proj, projT, 128, 128);

  // qkv = hs @ qkv_w : M=4096, N=6144, K=4096 -> f16
  gemm_f16_wmma<4096, 6144, 4096, false><<<dim3(6144 / 128, 4096 / 128), dim3(256), 0,
                                           stream>>>(hs16, qkvwT, (void*)qkv16);

  // fused per-head DCT attention (B*NH = 1024 blocks)
  dct_attn_kernel<<<dim3(1024), dim3(256), 0, stream>>>(qkv16, projT, wsq, attn16);

  // out = attn @ o_w : M=4096, N=4096, K=4096 -> f32 into d_out
  gemm_f16_wmma<4096, 4096, 4096, true><<<dim3(4096 / 128, 4096 / 128), dim3(256), 0,
                                          stream>>>(attn16, owT, d_out);
}
